// MLAAttention_86217173500119
// MI455X (gfx1250) — matrix operand order repaired and checked by reading, not compiled, not run on hardware
//
#include <hip/hip_runtime.h>
#include <hip/hip_bf16.h>

typedef __bf16 v16bf __attribute__((ext_vector_type(16)));
typedef __bf16 v8bf  __attribute__((ext_vector_type(8)));
typedef float  v8f   __attribute__((ext_vector_type(8)));
typedef int    v4i   __attribute__((ext_vector_type(4)));

union V16 { v16bf v; v8bf h[2]; };

#define WMMA_BF16(a, b, c) \
  __builtin_amdgcn_wmma_f32_16x16x32_bf16(false, (a), false, (b), (short)0, (c), false, false)

constexpr int BATCH = 2, SEQ = 2048, HID = 2048, NHEAD = 16, LATD = 512, HD = 128;
constexpr int MROWS = BATCH * SEQ;             // 4096
constexpr float SCALE = 0.08838834764831845f;  // 128^-0.5

// --- CDNA5 async global->LDS copies (ASYNCcnt-tracked), guarded -------------
#if defined(__HIP_DEVICE_COMPILE__) && __has_builtin(__builtin_amdgcn_global_load_async_to_lds_b128)
#define HAVE_ASYNC 1
#else
#define HAVE_ASYNC 0
#endif

#if HAVE_ASYNC
__device__ __forceinline__ void async_cp16(const __bf16* g, __bf16* l) {
  // builtin expects v4i* params (b128 = int4); generic->AS1/AS3 handled by clang's
  // builtin address-space rewrite.
  __builtin_amdgcn_global_load_async_to_lds_b128(
      (v4i*)(void*)const_cast<__bf16*>(g), (v4i*)(void*)l, 0, 0);
}
__device__ __forceinline__ void wait_async0() {
#if __has_builtin(__builtin_amdgcn_s_wait_asynccnt)
  __builtin_amdgcn_s_wait_asynccnt(0);
#else
  asm volatile("s_wait_asynccnt 0x0" ::: "memory");
#endif
}
#endif

// ---------------------------------------------------------------------------
// fp32 -> bf16 conversion
__global__ void cvt_bf16_kernel(const float* __restrict__ in, __bf16* __restrict__ out, int n) {
  for (int i = blockIdx.x * blockDim.x + threadIdx.x; i < n; i += gridDim.x * blockDim.x)
    out[i] = (__bf16)in[i];
}

// out[n*K + k] = bf16(a[k*N + n] + (b ? b[k*N + n] : 0))  (transpose + optional rope fold)
__global__ void cvt_addT_kernel(const float* __restrict__ a, const float* __restrict__ b,
                                __bf16* __restrict__ outT, int K, int N) {
  int total = K * N;
  for (int i = blockIdx.x * blockDim.x + threadIdx.x; i < total; i += gridDim.x * blockDim.x) {
    int k = i / N, n = i - k * N;
    float v = a[i];
    if (b) v += b[i];
    outT[(size_t)n * K + k] = (__bf16)v;
  }
}

// ---------------------------------------------------------------------------
// bf16 WMMA GEMM: C[m][n] = sum_k A[m][k] * BT[n][k]
// Block tile 256x128, 256 threads = 8 waves (4x2), wave tile 64x64, BK=32,
// double-buffered LDS, async global->LDS when available.
__global__ __launch_bounds__(256) void gemm_bf16_wmma(
    const __bf16* __restrict__ A, int lda,
    const __bf16* __restrict__ BT,   // [N][K] row-major (pre-transposed weights)
    float* __restrict__ Cf, __bf16* __restrict__ Cb, int ldc, int K) {
  __shared__ __bf16 Alds[2][256 * 40];
  __shared__ __bf16 Blds[2][128 * 40];

  const int tid = threadIdx.x;
  const int lane = tid & 31, w = tid >> 5;
  const int wm = w >> 1, wn = w & 1;   // 4 x 2 wave grid, wave tile 64x64
  const int r = lane & 15, kh = lane >> 4;
  const int m0 = blockIdx.y * 256, n0 = blockIdx.x * 128;

  v8f zf = {0.f, 0.f, 0.f, 0.f, 0.f, 0.f, 0.f, 0.f};
  v8f acc[4][4];
#pragma unroll
  for (int mt = 0; mt < 4; ++mt)
#pragma unroll
    for (int nt = 0; nt < 4; ++nt) acc[mt][nt] = zf;

  // loader mapping: A 256 rows x 32 (1 thread/row), B 128 rows x 32 (2 threads/row)
  const int arow = tid;
  const int brow = tid >> 1, bkoff = (tid & 1) * 16;
  const __bf16* Ag = A + (size_t)(m0 + arow) * lda;
  const __bf16* Bg = BT + (size_t)(n0 + brow) * K + bkoff;

  const int nk = K >> 5;

  // prologue: stage tile 0 into buffer 0
#if HAVE_ASYNC
#pragma unroll
  for (int i = 0; i < 4; ++i) async_cp16(Ag + i * 8, &Alds[0][arow * 40 + i * 8]);
#pragma unroll
  for (int i = 0; i < 2; ++i) async_cp16(Bg + i * 8, &Blds[0][brow * 40 + bkoff + i * 8]);
#else
  {
    v8bf av[4], bv[2];
#pragma unroll
    for (int i = 0; i < 4; ++i) av[i] = *(const v8bf*)(Ag + i * 8);
#pragma unroll
    for (int i = 0; i < 2; ++i) bv[i] = *(const v8bf*)(Bg + i * 8);
#pragma unroll
    for (int i = 0; i < 4; ++i) *(v8bf*)&Alds[0][arow * 40 + i * 8] = av[i];
#pragma unroll
    for (int i = 0; i < 2; ++i) *(v8bf*)&Blds[0][brow * 40 + bkoff + i * 8] = bv[i];
  }
#endif

  for (int it = 0; it < nk; ++it) {
    const int cur = it & 1;
#if HAVE_ASYNC
    wait_async0();
#endif
    __syncthreads();

#if HAVE_ASYNC
    if (it + 1 < nk) {
      const int nb = cur ^ 1, k0 = (it + 1) << 5;
#pragma unroll
      for (int i = 0; i < 4; ++i) async_cp16(Ag + k0 + i * 8, &Alds[nb][arow * 40 + i * 8]);
#pragma unroll
      for (int i = 0; i < 2; ++i) async_cp16(Bg + k0 + i * 8, &Blds[nb][brow * 40 + bkoff + i * 8]);
    }
#else
    v8bf av[4], bv[2];
    const bool more = (it + 1 < nk);
    if (more) {
      const int k0 = (it + 1) << 5;
#pragma unroll
      for (int i = 0; i < 4; ++i) av[i] = *(const v8bf*)(Ag + k0 + i * 8);
#pragma unroll
      for (int i = 0; i < 2; ++i) bv[i] = *(const v8bf*)(Bg + k0 + i * 8);
    }
#endif

    // compute from buffer `cur`
    V16 bf[4];
#pragma unroll
    for (int nt = 0; nt < 4; ++nt) {
      int c = wn * 64 + nt * 16 + r;
      bf[nt].h[0] = *(const v8bf*)&Blds[cur][c * 40 + kh * 8];
      bf[nt].h[1] = *(const v8bf*)&Blds[cur][c * 40 + kh * 8 + 16];
    }
#pragma unroll
    for (int mt = 0; mt < 4; ++mt) {
      int rr = wm * 64 + mt * 16 + r;
      V16 af;
      af.h[0] = *(const v8bf*)&Alds[cur][rr * 40 + kh * 8];        // K = kh*8 .. +7
      af.h[1] = *(const v8bf*)&Alds[cur][rr * 40 + kh * 8 + 16];   // K = kh*8+16 .. +23
#pragma unroll
      for (int nt = 0; nt < 4; ++nt) acc[mt][nt] = WMMA_BF16(af.v, bf[nt].v, acc[mt][nt]);
    }

#if !HAVE_ASYNC
    if (more) {
      const int nb = cur ^ 1;
#pragma unroll
      for (int i = 0; i < 4; ++i) *(v8bf*)&Alds[nb][arow * 40 + i * 8] = av[i];
#pragma unroll
      for (int i = 0; i < 2; ++i) *(v8bf*)&Blds[nb][brow * 40 + bkoff + i * 8] = bv[i];
    }
#endif
  }

#pragma unroll
  for (int mt = 0; mt < 4; ++mt)
#pragma unroll
    for (int nt = 0; nt < 4; ++nt)
#pragma unroll
      for (int j = 0; j < 8; ++j) {
        int row = m0 + wm * 64 + mt * 16 + j + 8 * kh;
        int col = n0 + wn * 64 + nt * 16 + r;
        if (Cf) Cf[(size_t)row * ldc + col] = acc[mt][nt][j];
        else    Cb[(size_t)row * ldc + col] = (__bf16)acc[mt][nt][j];
      }
}

// ---------------------------------------------------------------------------
// Flash attention, causal, per (b, h, 128-query block). 8 waves x 16 rows.
__global__ __launch_bounds__(256) void mla_attn_kernel(
    const __bf16* __restrict__ Q, const __bf16* __restrict__ Km,
    const __bf16* __restrict__ Vm, __bf16* __restrict__ AO) {
  __shared__ __bf16 Klds[64 * 136];    // [key][d]
  __shared__ __bf16 Vlds[128 * 72];    // [d][key] (transposed)
  __shared__ __bf16 Plds[8 * 16 * 64]; // per-wave P tiles [row][key]

  const int tid = threadIdx.x, lane = tid & 31, w = tid >> 5;
  const int r = lane & 15, kh = lane >> 4;
  const int qb = blockIdx.x, h = blockIdx.y, b = blockIdx.z;
  const int s0w = qb * 128 + w * 16;
  const size_t hoff = (size_t)h * HD;

  // Q fragments (A-layout), read directly from global, row = s0w + r
  V16 qf[4];
  {
    const __bf16* qp = Q + ((size_t)(b * SEQ + s0w + r)) * HID + hoff;
#pragma unroll
    for (int kk = 0; kk < 4; ++kk) {
      qf[kk].h[0] = *(const v8bf*)(qp + kk * 32 + kh * 8);
      qf[kk].h[1] = *(const v8bf*)(qp + kk * 32 + kh * 8 + 16);
    }
  }

  v8f zf = {0.f, 0.f, 0.f, 0.f, 0.f, 0.f, 0.f, 0.f};
  v8f o[8];
  float mrow[8], lrow[8];
#pragma unroll
  for (int i = 0; i < 8; ++i) { o[i] = zf; mrow[i] = -3e38f; lrow[i] = 0.f; }

  const int kr = tid >> 2, doff = (tid & 3) * 32;
  __bf16* pw = &Plds[w * 16 * 64];

  const int nkb = 2 * qb + 2;  // key blocks of 64 covering causal range of this block
  for (int kb = 0; kb < nkb; ++kb) {
    const int k0 = kb * 64;
    const __bf16* kp = Km + ((size_t)(b * SEQ + k0 + kr)) * HID + hoff + doff;
    const __bf16* vp = Vm + ((size_t)(b * SEQ + k0 + kr)) * HID + hoff + doff;
#if HAVE_ASYNC
    // K tile: direct async copy into [key][d] LDS layout
#pragma unroll
    for (int i = 0; i < 4; ++i) async_cp16(kp + i * 8, &Klds[kr * 136 + doff + i * 8]);
    // V tile: manual transpose into [d][key]
    v8bf vv[4];
#pragma unroll
    for (int i = 0; i < 4; ++i) vv[i] = *(const v8bf*)(vp + i * 8);
#pragma unroll
    for (int i = 0; i < 4; ++i)
#pragma unroll
      for (int e = 0; e < 8; ++e) Vlds[(doff + i * 8 + e) * 72 + kr] = vv[i][e];
    wait_async0();
#else
    v8bf kv[4], vv[4];
#pragma unroll
    for (int i = 0; i < 4; ++i) { kv[i] = *(const v8bf*)(kp + i * 8); vv[i] = *(const v8bf*)(vp + i * 8); }
#pragma unroll
    for (int i = 0; i < 4; ++i) *(v8bf*)&Klds[kr * 136 + doff + i * 8] = kv[i];
#pragma unroll
    for (int i = 0; i < 4; ++i)
#pragma unroll
      for (int e = 0; e < 8; ++e) Vlds[(doff + i * 8 + e) * 72 + kr] = vv[i][e];
#endif
    __syncthreads();

    const bool active = (k0 <= s0w + 15);  // wave-uniform: EXEC stays all-ones
    v8f sc[4];
    if (active) {
#pragma unroll
      for (int nt = 0; nt < 4; ++nt) sc[nt] = zf;
      // S = Q @ K^T : contraction over d (4 steps of 32)
#pragma unroll
      for (int kk = 0; kk < 4; ++kk)
#pragma unroll
        for (int nt = 0; nt < 4; ++nt) {
          V16 bfr;
          bfr.h[0] = *(const v8bf*)&Klds[(nt * 16 + r) * 136 + kk * 32 + kh * 8];
          bfr.h[1] = *(const v8bf*)&Klds[(nt * 16 + r) * 136 + kk * 32 + kh * 8 + 16];
          sc[nt] = WMMA_BF16(qf[kk].v, bfr.v, sc[nt]);
        }
      // causal mask + online softmax (rows j+8*kh, cols striped over 16 lanes)
#pragma unroll
      for (int j = 0; j < 8; ++j) {
        const int qrow = s0w + j + 8 * kh;
        float mj = -3e38f;
#pragma unroll
        for (int nt = 0; nt < 4; ++nt) {
          int key = k0 + nt * 16 + r;
          float v = sc[nt][j] * SCALE + (key <= qrow ? 0.f : -1e9f);
          sc[nt][j] = v;
          mj = fmaxf(mj, v);
        }
#pragma unroll
        for (int m = 1; m < 16; m <<= 1) mj = fmaxf(mj, __shfl_xor(mj, m, 32));
        float mnew = fmaxf(mrow[j], mj);
        float scl = __expf(mrow[j] - mnew);
        float rs = 0.f;
#pragma unroll
        for (int nt = 0; nt < 4; ++nt) {
          float p = __expf(sc[nt][j] - mnew);
          sc[nt][j] = p;
          rs += p;
        }
#pragma unroll
        for (int m = 1; m < 16; m <<= 1) rs += __shfl_xor(rs, m, 32);
        lrow[j] = lrow[j] * scl + rs;
        mrow[j] = mnew;
#pragma unroll
        for (int ot = 0; ot < 8; ++ot) o[ot][j] *= scl;
      }
      // stage P (bf16) into A-fragment-friendly [row][key] layout
#pragma unroll
      for (int nt = 0; nt < 4; ++nt)
#pragma unroll
        for (int j = 0; j < 8; ++j)
          pw[(j + 8 * kh) * 64 + nt * 16 + r] = (__bf16)sc[nt][j];
    }
    __syncthreads();

    if (active) {
      // O += P @ V : contraction over keys (2 steps of 32)
#pragma unroll
      for (int kk = 0; kk < 2; ++kk) {
        V16 pa;
        pa.h[0] = *(const v8bf*)&pw[r * 64 + kk * 32 + kh * 8];
        pa.h[1] = *(const v8bf*)&pw[r * 64 + kk * 32 + kh * 8 + 16];
#pragma unroll
        for (int ot = 0; ot < 8; ++ot) {
          V16 vb;
          vb.h[0] = *(const v8bf*)&Vlds[(ot * 16 + r) * 72 + kk * 32 + kh * 8];
          vb.h[1] = *(const v8bf*)&Vlds[(ot * 16 + r) * 72 + kk * 32 + kh * 8 + 16];
          o[ot] = WMMA_BF16(pa.v, vb.v, o[ot]);
        }
      }
    }
    __syncthreads();
  }

  // normalize + write (b, s, h*128 + d) bf16
#pragma unroll
  for (int ot = 0; ot < 8; ++ot)
#pragma unroll
    for (int j = 0; j < 8; ++j) {
      int srow = s0w + j + 8 * kh;
      AO[((size_t)(b * SEQ + srow)) * HID + hoff + ot * 16 + r] = (__bf16)(o[ot][j] / lrow[j]);
    }
}

// ---------------------------------------------------------------------------
extern "C" void kernel_launch(void* const* d_in, const int* in_sizes, int n_in,
                              void* d_out, int out_size, void* d_ws, size_t ws_size,
                              hipStream_t stream) {
  (void)in_sizes; (void)n_in; (void)out_size; (void)ws_size;

  const float* hs      = (const float*)d_in[0];
  // d_in[1] attention_mask, d_in[2] position_ids: causal mask recomputed in-kernel
  const float* Wq      = (const float*)d_in[3];
  const float* Wkv     = (const float*)d_in[4];
  const float* Wk_up   = (const float*)d_in[5];
  const float* Wv_up   = (const float*)d_in[6];
  const float* Wq_rope = (const float*)d_in[7];
  const float* Wk_rope = (const float*)d_in[8];
  const float* Wo      = (const float*)d_in[9];
  float* out = (float*)d_out;

  char* p = (char*)d_ws;
  auto alloc = [&](size_t elems) -> __bf16* {
    __bf16* r = (__bf16*)p;
    p += (elems * 2 + 255) & ~(size_t)255;
    return r;
  };
  __bf16* hbf  = alloc((size_t)MROWS * HID);       // hidden bf16
  __bf16* WqT  = alloc((size_t)HID * HID);         // (Wq + Wq_rope)^T   [N=2048][K=2048]
  __bf16* WkvT = alloc((size_t)1024 * HID);        // Wkv^T              [N=1024][K=2048]
  __bf16* WkT  = alloc((size_t)HID * LATD);        // (Wk_up+Wk_rope)^T  [N=2048][K=512]
  __bf16* WvT  = alloc((size_t)HID * LATD);        // Wv_up^T            [N=2048][K=512]
  __bf16* WoT  = alloc((size_t)HID * HID);         // Wo^T               [N=2048][K=2048]
  __bf16* Qb   = alloc((size_t)MROWS * HID);
  __bf16* KVb  = alloc((size_t)MROWS * 1024);
  __bf16* Kb   = alloc((size_t)MROWS * HID);
  __bf16* Vb   = alloc((size_t)MROWS * HID);
  __bf16* AOb  = alloc((size_t)MROWS * HID);

  cvt_bf16_kernel<<<2048, 256, 0, stream>>>(hs, hbf, MROWS * HID);
  cvt_addT_kernel<<<2048, 256, 0, stream>>>(Wq, Wq_rope, WqT, HID, HID);
  cvt_addT_kernel<<<1024, 256, 0, stream>>>(Wkv, nullptr, WkvT, HID, 1024);
  cvt_addT_kernel<<<1024, 256, 0, stream>>>(Wk_up, Wk_rope, WkT, LATD, HID);
  cvt_addT_kernel<<<1024, 256, 0, stream>>>(Wv_up, nullptr, WvT, LATD, HID);
  cvt_addT_kernel<<<2048, 256, 0, stream>>>(Wo, nullptr, WoT, HID, HID);

  dim3 blk(256);
  // Q = hidden @ (Wq + Wq_rope)
  gemm_bf16_wmma<<<dim3(HID / 128, MROWS / 256), blk, 0, stream>>>(hbf, HID, WqT, nullptr, Qb, HID, HID);
  // KV = hidden @ Wkv
  gemm_bf16_wmma<<<dim3(1024 / 128, MROWS / 256), blk, 0, stream>>>(hbf, HID, WkvT, nullptr, KVb, 1024, HID);
  // K = k_latent @ (Wk_up + Wk_rope)
  gemm_bf16_wmma<<<dim3(HID / 128, MROWS / 256), blk, 0, stream>>>(KVb, 1024, WkT, nullptr, Kb, HID, LATD);
  // V = v_latent @ Wv_up
  gemm_bf16_wmma<<<dim3(HID / 128, MROWS / 256), blk, 0, stream>>>(KVb + LATD, 1024, WvT, nullptr, Vb, HID, LATD);
  // flash attention
  mla_attn_kernel<<<dim3(SEQ / 128, NHEAD, BATCH), blk, 0, stream>>>(Qb, Kb, Vb, AOb);
  // out = attn_out @ Wo (fp32 output)
  gemm_bf16_wmma<<<dim3(HID / 128, MROWS / 256), blk, 0, stream>>>(AOb, HID, WoT, out, nullptr, HID, HID);
}
